// SingleFrameFlowTransformer_18236431139337
// MI455X (gfx1250) — matrix operand order repaired and checked by reading, not compiled, not run on hardware
//
#include <hip/hip_runtime.h>
#include <hip/hip_bf16.h>

#define BT   128
#define NJ   27
#define DD   512
#define NHD  8
#define NLAY 6
#define NFF  2048
#define NPTS 1024
#define NTOK (BT * NJ)     // 3456
#define NEGB (-1.0e9f)
#define KC   128           // GEMM k-chunk staged via TDM

typedef __attribute__((ext_vector_type(16))) __bf16 v16bf;
typedef __attribute__((ext_vector_type(8)))  __bf16 v8bf;
typedef __attribute__((ext_vector_type(8)))  float  v8f;
typedef __attribute__((ext_vector_type(4)))  unsigned u32x4;
typedef __attribute__((ext_vector_type(8)))  unsigned u32x8;

__constant__ int   c_parent[NJ] = {0,0,1,2,3,4,5,6,7,8,8,3,11,12,13,14,15,15,0,18,19,20,0,22,23,24,3};
__constant__ float c_rscale[NJ] = {0.6f,0.6f,0.6f,0.6f,1.0f,1.0f,1.0f,1.0f,1.0f,1.5f,1.5f,1.0f,1.0f,
                                   1.0f,1.0f,1.0f,1.5f,1.5f,0.6f,1.0f,1.5f,1.5f,0.6f,1.0f,1.5f,1.5f,1.5f};

// ---------------- WMMA fragment helpers (wave32 bf16 16x16x32 layouts) ----------------
// A (16x32, 16-bit): lanes 0-15 row m=l, elems 0-7 = K kb+0..7, elems 8-15 = K kb+16..23
//                    lanes 16-31 same rows, K kb+8..15 / kb+24..31
__device__ __forceinline__ v16bf frag_a(const __bf16* p) {
  v8bf lo = *(const v8bf*)(p);
  v8bf hi = *(const v8bf*)(p + 16);
  v16bf r;
#pragma unroll
  for (int i = 0; i < 8; ++i) { r[i] = lo[i]; r[i + 8] = hi[i]; }
  return r;
}
// B (32x16, 16-bit): lane l holds column n=l&15; lanes 0-15 K kb+0..7 / kb+16..23, lanes 16-31 K kb+8..15 / kb+24..31
__device__ __forceinline__ v16bf frag_b(const __bf16* p) {
  v8bf lo = *(const v8bf*)(p);
  v8bf hi = *(const v8bf*)(p + 16);
  v16bf r;
#pragma unroll
  for (int i = 0; i < 8; ++i) { r[i] = lo[i]; r[i + 8] = hi[i]; }
  return r;
}
__device__ __forceinline__ v16bf zero16() {
  v16bf r;
#pragma unroll
  for (int i = 0; i < 16; ++i) r[i] = (__bf16)0.0f;
  return r;
}
#define WMMA_BF16(a, b, c) __builtin_amdgcn_wmma_f32_16x16x32_bf16(false, (a), false, (b), (short)0, (c), false, false)

// ---------------- Tensor Data Mover: DMA a 64-row x KC-col bf16 tile into LDS ----------------
// D# per CDNA5 ISA ch.8: group0 = {count/flags, lds_addr, global_addr[31:0], global_addr[56:32]|type<<30}
// group1 = {mask/data_size, tensor_dim0 fields, tensor_dim1/tile_dim0, tile_dim1/2, strides...}
__device__ __forceinline__ void tdm_load_tile64xKC(unsigned ldsOff, const __bf16* gsrc, unsigned ldwElems) {
  const unsigned long long ga = (unsigned long long)gsrc;
  u32x4 g0;
  g0[0] = 1u;                                            // count=1 (valid user descriptor)
  g0[1] = ldsOff;                                        // LDS byte address
  g0[2] = (unsigned)(ga & 0xffffffffu);                  // global_addr[31:0]
  g0[3] = (unsigned)((ga >> 32) & 0x1ffffffu) | (2u << 30); // global_addr[56:32] | type=2
  u32x8 g1;
  g1[0] = 0x00010000u;                                   // workgroup_mask=0, data_size=1 (2 bytes)
  g1[1] = ((unsigned)KC << 16);                          // tensor_dim0[15:0]=KC  (bits 79:48)
  g1[2] = (64u << 16);                                   // tensor_dim0[31:16]=0, tensor_dim1[15:0]=64
  g1[3] = ((unsigned)KC << 16);                          // tensor_dim1[31:16]=0, tile_dim0=KC
  g1[4] = 64u;                                           // tile_dim1=64, tile_dim2=0
  g1[5] = ldwElems;                                      // tensor_dim0_stride[31:0] (elements)
  g1[6] = 0u;                                            // stride[47:32]=0, tensor_dim1_stride lo=0
  g1[7] = 0u;
  asm volatile("tensor_load_to_lds %0, %1" :: "s"(g0), "s"(g1) : "memory");
}

// ---------------- generic bf16 WMMA GEMM: C[M,N] = A[M,K] @ W[N,K]^T (+bias) ----------------
// block = 256 threads = 8 waves; block tile 128(M) x 64(N); wave tile 16 x 64 (4 n-tiles).
// W tile (64 x KC per chunk) is DMA'd into LDS by the TDM, double buffered (TENSORcnt).
// A fragments stream straight from global (rows are wave-private; resident in L2).
// EPI: 0 outF=gemm+bias  1 outF+=gemm+bias  2 outB=bf16(gemm+bias)  3 outB=bf16(relu)
//      4 KV split (c<512 -> K row-major bf16; c>=512 -> Vt transposed bf16)
//      5 outF = gemm * rowscale[m]
template<int EPI>
__global__ __launch_bounds__(256) void gemm_bf16_kernel(
    const __bf16* __restrict__ A, int lda, long long sA,
    const __bf16* __restrict__ W, int ldw, long long sW,
    const float*  __restrict__ bias,
    float*        __restrict__ outF, long long sOF,
    __bf16*       __restrict__ outB, long long sOB,
    __bf16*       __restrict__ out2,
    const float*  __restrict__ rowscale, long long sRS,
    int M, int N, int K, int ldo)
{
  __shared__ __bf16 sWt[2][64 * KC];                     // 32 KB double buffer

  const int z = blockIdx.z;
  A += (long long)z * sA;
  W += (long long)z * sW;
  if (EPI == 0 || EPI == 1 || EPI == 5) outF += (long long)z * sOF;
  if (EPI == 2 || EPI == 3 || EPI == 4) outB += (long long)z * sOB;
  if (EPI == 5) rowscale += (long long)z * sRS;

  const int wave = threadIdx.x >> 5;
  const int lane = threadIdx.x & 31;
  const int ln16 = lane & 15;
  const int kh8  = (lane >> 4) * 8;
  const int kh16 = (lane >> 4) * 16;

  const int rowBase = blockIdx.x * 128 + wave * 16;
  const int colBase = blockIdx.y * 64;
  const int r = rowBase + ln16;
  const bool rok = (r < M);

  const __bf16* Wtile = W + (long long)colBase * ldw;    // 64 consecutive W rows
  const unsigned lds0 = (unsigned)(unsigned long long)(void*)&sWt[0][0];
  const unsigned lds1 = (unsigned)(unsigned long long)(void*)&sWt[1][0];

  v8f acc[4];
#pragma unroll
  for (int t = 0; t < 4; ++t)
#pragma unroll
    for (int i = 0; i < 8; ++i) acc[t][i] = 0.0f;

  const int nCh = K / KC;
  if (wave == 0) tdm_load_tile64xKC(lds0, Wtile, (unsigned)ldw);

  for (int c = 0; c < nCh; ++c) {
    if (wave == 0) {
      if (c + 1 < nCh) {
        tdm_load_tile64xKC((c & 1) ? lds0 : lds1, Wtile + (long long)(c + 1) * KC, (unsigned)ldw);
        __builtin_amdgcn_s_wait_tensorcnt(1);            // chunk c complete, c+1 in flight
      } else {
        __builtin_amdgcn_s_wait_tensorcnt(0);
      }
    }
    __syncthreads();
    const __bf16* sWc = sWt[c & 1];
#pragma unroll
    for (int kk = 0; kk < KC; kk += 32) {
      const int kb = c * KC + kk;
      v16bf a;
      if (rok) a = frag_a(A + (long long)r * lda + kb + kh8);
      else     a = zero16();
#pragma unroll
      for (int t = 0; t < 4; ++t) {
        v16bf wv = frag_b(sWc + (t * 16 + ln16) * KC + kk + kh8);
        acc[t] = WMMA_BF16(a, wv, acc[t]);
      }
    }
    __syncthreads();
  }

  const int m0 = rowBase + ((lane < 16) ? 0 : 8);
#pragma unroll
  for (int t = 0; t < 4; ++t) {
    const int c = colBase + t * 16 + ln16;
    const float bv = bias ? bias[c] : 0.0f;
#pragma unroll
    for (int rr = 0; rr < 8; ++rr) {
      const int m = m0 + rr;
      if (m >= M) continue;
      const float v = acc[t][rr] + bv;
      if (EPI == 0) {
        outF[(long long)m * ldo + c] = v;
      } else if (EPI == 1) {
        outF[(long long)m * ldo + c] += v;
      } else if (EPI == 2) {
        outB[(long long)m * ldo + c] = (__bf16)v;
      } else if (EPI == 3) {
        outB[(long long)m * ldo + c] = (__bf16)fmaxf(v, 0.0f);
      } else if (EPI == 4) {
        if (c < 512) {
          outB[(long long)m * 512 + c] = (__bf16)v;                       // K[b*1024+n][d]
        } else {
          const int bb = m >> 10, n = m & 1023;
          out2[((long long)bb * 512 + (c - 512)) * 1024 + n] = (__bf16)v; // Vt[b][d][n]
        }
      } else if (EPI == 5) {
        outF[(long long)m * ldo + c] = v * rowscale[m];
      }
    }
  }
}

// ---------------- fused cross-attention per (b,h): WMMA scores -> LDS softmax -> WMMA AV ----------------
__global__ __launch_bounds__(256) void ca_attn_kernel(
    const __bf16* __restrict__ Qb, const __bf16* __restrict__ Kb,
    const __bf16* __restrict__ Vt, const float* __restrict__ maskBias,
    __bf16* __restrict__ Ob)
{
  extern __shared__ char smem[];
  float*  s_sc  = (float*)smem;                              // 32*1024 f32 (128 KB)
  __bf16* s_pb  = (__bf16*)(smem + 32 * 1024 * 4);           // 32*1024 bf16 (64 KB)
  float*  s_red = (float*)(smem + 32 * 1024 * 4 + 32 * 1024 * 2); // 32*8 f32

  const int b = blockIdx.x >> 3;
  const int h = blockIdx.x & 7;
  const int wave = threadIdx.x >> 5, lane = threadIdx.x & 31;
  const int ln16 = lane & 15;
  const int kh8 = (lane >> 4) * 8, kh16 = (lane >> 4) * 16;

  const __bf16* Qbase = Qb + ((long long)b * NJ) * DD + h * 64;
  const __bf16* Kbase = Kb + ((long long)b * NPTS) * DD + h * 64;

  // scores S[32][1024] = Q @ K^T * 0.125 + mask bias
  for (int nt = wave; nt < 64; nt += 8) {
    const int key = nt * 16 + ln16;
#pragma unroll
    for (int mt = 0; mt < 2; ++mt) {
      const int q = mt * 16 + ln16;
      v8f acc;
#pragma unroll
      for (int i = 0; i < 8; ++i) acc[i] = 0.0f;
#pragma unroll
      for (int kk = 0; kk < 2; ++kk) {
        const int kb = kk * 32;
        v16bf a;
        if (q < NJ) a = frag_a(Qbase + (long long)q * DD + kb + kh8);
        else        a = zero16();
        v16bf bv = frag_b(Kbase + (long long)key * DD + kb + kh8);
        acc = WMMA_BF16(a, bv, acc);
      }
      const int m0 = mt * 16 + ((lane < 16) ? 0 : 8);
      const int col = nt * 16 + ln16;
#pragma unroll
      for (int rr = 0; rr < 8; ++rr) {
        const int m = m0 + rr;
        float s = 0.0f;
        if (m < NJ) s = acc[rr] * 0.125f + maskBias[((long long)b * NJ + m) * NPTS + col];
        s_sc[m * 1024 + col] = s;
      }
    }
  }
  __syncthreads();
  // softmax per row: 8 threads per row
  {
    const int row = threadIdx.x >> 3, sub = threadIdx.x & 7;
    float mx = -3.4e38f;
    for (int c = sub; c < 1024; c += 8) mx = fmaxf(mx, s_sc[row * 1024 + c]);
    s_red[row * 8 + sub] = mx;
    __syncthreads();
    float rmx = s_red[row * 8];
#pragma unroll
    for (int i = 1; i < 8; ++i) rmx = fmaxf(rmx, s_red[row * 8 + i]);
    __syncthreads();
    float sum = 0.0f;
    for (int c = sub; c < 1024; c += 8) {
      float e = __expf(s_sc[row * 1024 + c] - rmx);
      s_sc[row * 1024 + c] = e;
      sum += e;
    }
    s_red[row * 8 + sub] = sum;
    __syncthreads();
    float tot = 0.0f;
#pragma unroll
    for (int i = 0; i < 8; ++i) tot += s_red[row * 8 + i];
    const float inv = 1.0f / tot;
    for (int c = sub; c < 1024; c += 8)
      s_pb[row * 1024 + c] = (__bf16)(s_sc[row * 1024 + c] * inv);
  }
  __syncthreads();
  // O[32][64] = P @ V  (B-fragments contiguous thanks to transposed Vt)
  const int mt = wave & 1, nt = wave >> 1;
  const __bf16* Vbase = Vt + ((long long)b * DD + h * 64 + nt * 16 + ln16) * NPTS;
  const int prow = mt * 16 + ln16;
  v8f acc;
#pragma unroll
  for (int i = 0; i < 8; ++i) acc[i] = 0.0f;
  for (int kb = 0; kb < 1024; kb += 32) {
    v16bf a  = frag_a(s_pb + prow * 1024 + kb + kh8);
    v16bf bv = frag_b(Vbase + kb + kh8);
    acc = WMMA_BF16(a, bv, acc);
  }
  const int m0 = mt * 16 + ((lane < 16) ? 0 : 8);
  const int col = h * 64 + nt * 16 + ln16;
#pragma unroll
  for (int rr = 0; rr < 8; ++rr) {
    const int q = m0 + rr;
    if (q < NJ) Ob[((long long)b * NJ + q) * DD + col] = (__bf16)acc[rr];
  }
}

// ---------------- self-attention (27x27, VALU, one wave per (b,h)) ----------------
__global__ __launch_bounds__(32) void sa_attn_kernel(const float* __restrict__ qkv, __bf16* __restrict__ Ob)
{
  __shared__ float sQ[NJ * 64];
  __shared__ float sK[NJ * 64];
  __shared__ float sV[NJ * 64];
  __shared__ float sSc[NJ * NJ];
  const int b = blockIdx.x >> 3, h = blockIdx.x & 7;
  const float* base = qkv + ((long long)b * NJ) * 1536;
  const int lane = threadIdx.x;
  for (int i = lane; i < NJ * 64; i += 32) {
    const int j = i >> 6, d = i & 63;
    sQ[i] = base[(long long)j * 1536 + h * 64 + d];
    sK[i] = base[(long long)j * 1536 + 512 + h * 64 + d];
    sV[i] = base[(long long)j * 1536 + 1024 + h * 64 + d];
  }
  __syncthreads();
  if (lane < NJ) {
    const int pq = c_parent[lane];
    float mx = -3.4e38f;
    for (int k = 0; k < NJ; ++k) {
      float s = 0.0f;
      for (int d = 0; d < 64; ++d) s += sQ[lane * 64 + d] * sK[k * 64 + d];
      s *= 0.125f;
      const bool ok = (k == lane) || (pq == k) || (c_parent[k] == lane);
      s += ok ? 0.0f : NEGB;
      sSc[lane * NJ + k] = s;
      mx = fmaxf(mx, s);
    }
    float sum = 0.0f;
    for (int k = 0; k < NJ; ++k) {
      float e = __expf(sSc[lane * NJ + k] - mx);
      sSc[lane * NJ + k] = e;
      sum += e;
    }
    const float inv = 1.0f / sum;
    for (int d = 0; d < 64; ++d) {
      float o = 0.0f;
      for (int k = 0; k < NJ; ++k) o += sSc[lane * NJ + k] * sV[k * 64 + d];
      Ob[((long long)b * NJ + lane) * DD + h * 64 + d] = (__bf16)(o * inv);
    }
  }
}

// ---------------- LayerNorm: one wave per token, output bf16 ----------------
__global__ __launch_bounds__(256) void ln_kernel(const float* __restrict__ x,
                                                 const float* __restrict__ g,
                                                 const float* __restrict__ bta,
                                                 __bf16* __restrict__ out, int ntok)
{
  const int wave = threadIdx.x >> 5, lane = threadIdx.x & 31;
  const int tok = blockIdx.x * 8 + wave;
  if (tok >= ntok) return;
  const float* xr = x + (long long)tok * DD;
  float v[16];
  float s = 0.0f;
#pragma unroll
  for (int i = 0; i < 16; ++i) { v[i] = xr[lane + i * 32]; s += v[i]; }
#pragma unroll
  for (int m = 16; m > 0; m >>= 1) s += __shfl_xor(s, m, 32);
  const float mean = s * (1.0f / 512.0f);
  float vs = 0.0f;
#pragma unroll
  for (int i = 0; i < 16; ++i) { float d = v[i] - mean; vs += d * d; }
#pragma unroll
  for (int m = 16; m > 0; m >>= 1) vs += __shfl_xor(vs, m, 32);
  const float inv = rsqrtf(vs * (1.0f / 512.0f) + 1e-5f);
#pragma unroll
  for (int i = 0; i < 16; ++i) {
    const int c = lane + i * 32;
    out[(long long)tok * DD + c] = (__bf16)((v[i] - mean) * inv * g[c] + bta[c]);
  }
}

// ---------------- preprocessing ----------------
__global__ __launch_bounds__(256) void dw_mask_kernel(const float* __restrict__ x_t,
                                                      const float* __restrict__ pc_raw,
                                                      __bf16* __restrict__ dwb,
                                                      float* __restrict__ biasOut,
                                                      float* __restrict__ invDenom)
{
  __shared__ float sD2[NPTS];
  __shared__ float sSum[256];
  __shared__ float sMin[256];
  __shared__ int   sAI[256];
  const int b = blockIdx.x / NJ, j = blockIdx.x % NJ;
  const int tid = threadIdx.x;
  const float xt0 = x_t[((long long)b * NJ + j) * 3 + 0];
  const float xt1 = x_t[((long long)b * NJ + j) * 3 + 1];
  const float xt2 = x_t[((long long)b * NJ + j) * 3 + 2];
  const float r  = c_rscale[j] * 0.1f;
  const float r2 = r * r;
  const float inv2s2 = 2.0f / r2;   // 1/(2*(r/2)^2)
  float sum = 0.0f, mind2 = 3.4e38f;
  int mi = 0;
  for (int n = tid; n < NPTS; n += 256) {
    const float* p = pc_raw + ((long long)b * NPTS + n) * 6;
    const float dx = xt0 - p[0], dy = xt1 - p[1], dz = xt2 - p[2];
    const float d2 = dx * dx + dy * dy + dz * dz;
    sD2[n] = d2;
    const float w = __expf(-d2 * inv2s2);
    dwb[((long long)b * NJ + j) * NPTS + n] = (__bf16)w;
    sum += w;
    if (d2 < mind2) { mind2 = d2; mi = n; }
  }
  sSum[tid] = sum; sMin[tid] = mind2; sAI[tid] = mi;
  __syncthreads();
  for (int s = 128; s > 0; s >>= 1) {
    if (tid < s) {
      sSum[tid] += sSum[tid + s];
      const float mv = sMin[tid + s]; const int ai = sAI[tid + s];
      if (mv < sMin[tid] || (mv == sMin[tid] && ai < sAI[tid])) { sMin[tid] = mv; sAI[tid] = ai; }
    }
    __syncthreads();
  }
  const float tot = sSum[0];
  const bool allm = sMin[0] > r2;
  const int am = sAI[0];
  if (tid == 0) invDenom[b * NJ + j] = 1.0f / (tot + 1e-6f);
  for (int n = tid; n < NPTS; n += 256) {
    float bias;
    if (allm) bias = (n == am) ? 0.0f : NEGB;
    else      bias = (sD2[n] > r2) ? NEGB : 0.0f;
    biasOut[((long long)b * NJ + j) * NPTS + n] = bias;
  }
}

__global__ __launch_bounds__(256) void imp_kernel(const float* __restrict__ x_t,
                                                  const float* __restrict__ pc_raw,
                                                  float* __restrict__ imp)
{
  const long long idx = (long long)blockIdx.x * 256 + threadIdx.x;
  if (idx >= (long long)BT * NPTS) return;
  const int b = (int)(idx >> 10);
  const float* p = pc_raw + idx * 6;
  const float px = p[0], py = p[1], pz = p[2];
  float best = 0.0f;
  for (int j = 0; j < NJ; ++j) {
    const float* xt = x_t + ((long long)b * NJ + j) * 3;
    const float dx = xt[0] - px, dy = xt[1] - py, dz = xt[2] - pz;
    const float d2 = dx * dx + dy * dy + dz * dz;
    const float r = c_rscale[j] * 0.1f;
    best = fmaxf(best, __expf(-d2 * 2.0f / (r * r)));
  }
  imp[idx] = best;
}

__global__ __launch_bounds__(128) void memproj_kernel(const float* __restrict__ pc_raw,
                                                      const float* __restrict__ Wp,
                                                      const float* __restrict__ bp,
                                                      const float* __restrict__ imp,
                                                      __bf16* __restrict__ mem)
{
  const long long pt = blockIdx.x;
  const float* p = pc_raw + pt * 6;
  const float c0 = p[0], c1 = p[1], c2 = p[2], c3 = p[3], c4 = p[4], c5 = p[5];
  const float im = imp[pt];
  for (int d = threadIdx.x; d < DD; d += 128) {
    const float* w = Wp + (long long)d * 6;
    const float v = bp[d] + c0 * w[0] + c1 * w[1] + c2 * w[2] + c3 * w[3] + c4 * w[4] + c5 * w[5];
    mem[pt * DD + d] = (__bf16)(v * im);
  }
}

__global__ __launch_bounds__(64) void dop_kernel(const float* __restrict__ pc_raw,
                                                 const float* __restrict__ w1, const float* __restrict__ b1,
                                                 const float* __restrict__ w2, const float* __restrict__ b2,
                                                 __bf16* __restrict__ dophT)
{
  __shared__ float sh[64];
  const long long pt = blockIdx.x;
  const int b = (int)(pt >> 10), n = (int)(pt & 1023);
  const float dop = pc_raw[pt * 6 + 3];
  const int t = threadIdx.x;
  {
    const float z = dop * w1[t] + b1[t];
    sh[t] = z / (1.0f + __expf(-z));   // silu
  }
  __syncthreads();
  for (int d = t; d < DD; d += 64) {
    float v = b2[d];
    const float* w = w2 + (long long)d * 64;
    for (int k = 0; k < 64; ++k) v += sh[k] * w[k];
    dophT[((long long)b * DD + d) * NPTS + n] = (__bf16)v;   // transposed (B, D, N)
  }
}

__global__ __launch_bounds__(128) void jointproj_kernel(const float* __restrict__ x_t, const float* __restrict__ x0,
                                                        const float* __restrict__ jw, const float* __restrict__ jb,
                                                        const float* __restrict__ cw, const float* __restrict__ cb,
                                                        const float* __restrict__ dfw, const float* __restrict__ dfb,
                                                        const float* __restrict__ id_embed,
                                                        __bf16* __restrict__ cc, float* __restrict__ qacc)
{
  const int tok = blockIdx.x;
  const int b = tok / NJ, j = tok % NJ;
  const int p = c_parent[j];
  const float* xt = x_t + ((long long)b * NJ + j) * 3;
  const float* xp = x_t + ((long long)b * NJ + p) * 3;
  const float* xz = x0  + ((long long)b * NJ + j) * 3;
  const float a0 = xt[0], a1 = xt[1], a2 = xt[2];
  const float p0 = xp[0], p1 = xp[1], p2 = xp[2];
  const float z0 = xz[0], z1 = xz[1], z2 = xz[2];
  for (int d = threadIdx.x; d < DD; d += 128) {
    const float j0 = jw[d * 3], j1 = jw[d * 3 + 1], j2 = jw[d * 3 + 2];
    const float hx = jb[d] + j0 * a0 + j1 * a1 + j2 * a2;
    const float hp = jb[d] + j0 * p0 + j1 * p1 + j2 * p2;
    const float h0 = cb[d] + cw[d * 3] * z0 + cw[d * 3 + 1] * z1 + cw[d * 3 + 2] * z2;
    const float hd = dfb[d] + dfw[d * 3] * (a0 - z0) + dfw[d * 3 + 1] * (a1 - z1) + dfw[d * 3 + 2] * (a2 - z2);
    cc[(long long)tok * 1024 + d]       = (__bf16)hx;
    cc[(long long)tok * 1024 + 512 + d] = (__bf16)hp;
    qacc[(long long)tok * DD + d] = h0 + hd + id_embed[(long long)j * DD + d];
  }
}

__global__ __launch_bounds__(256) void time_kernel(const float* __restrict__ t,
                                                   const float* __restrict__ w1, const float* __restrict__ b1,
                                                   const float* __restrict__ w2, const float* __restrict__ b2,
                                                   float* __restrict__ ht)
{
  __shared__ float sh[DD];
  const int b = blockIdx.x;
  const float tv = t[b];
  for (int k = threadIdx.x; k < DD; k += 256) {
    const float z = tv * w1[k] + b1[k];
    sh[k] = z / (1.0f + __expf(-z));
  }
  __syncthreads();
  for (int d = threadIdx.x; d < DD; d += 256) {
    float v = b2[d];
    const float* w = w2 + (long long)d * DD;
    for (int k = 0; k < DD; ++k) v += sh[k] * w[k];
    ht[(long long)b * DD + d] = v;
  }
}

__global__ __launch_bounds__(256) void assemble_kernel(const float* __restrict__ qacc,
                                                       const float* __restrict__ hdop,
                                                       const float* __restrict__ ht,
                                                       float* __restrict__ x)
{
  const long long i = (long long)blockIdx.x * 256 + threadIdx.x;
  if (i >= (long long)NTOK * DD) return;
  const long long tok = i >> 9;
  const int d = (int)(i & 511);
  const int b = (int)(tok / NJ);
  x[i] = qacc[i] + hdop[i] + ht[(long long)b * DD + d];
}

__global__ __launch_bounds__(128) void vel_kernel(const float* __restrict__ x,
                                                  const float* __restrict__ vw,
                                                  const float* __restrict__ vb,
                                                  float* __restrict__ out)
{
  const int gw = (int)(((long long)blockIdx.x * 128 + threadIdx.x) >> 5);
  const int lane = threadIdx.x & 31;
  if (gw >= NTOK * 3) return;
  const int tok = gw / 3, c = gw % 3;
  const float* xr = x + (long long)tok * DD;
  const float* w = vw + (long long)c * DD;
  float s = 0.0f;
  for (int d = lane; d < DD; d += 32) s += xr[d] * w[d];
#pragma unroll
  for (int m = 16; m > 0; m >>= 1) s += __shfl_xor(s, m, 32);
  if (lane == 0) out[tok * 3 + c] = s + vb[c];
}

__global__ void f32_to_bf16_kernel(const float* __restrict__ s, __bf16* __restrict__ d, long long n) {
  long long i = (long long)blockIdx.x * blockDim.x + threadIdx.x;
  const long long stride = (long long)gridDim.x * blockDim.x;
  for (; i < n; i += stride) d[i] = (__bf16)s[i];
}

// ======================================================================================
extern "C" void kernel_launch(void* const* d_in, const int* in_sizes, int n_in,
                              void* d_out, int out_size, void* d_ws, size_t ws_size,
                              hipStream_t stream) {
  (void)in_sizes; (void)n_in; (void)out_size; (void)ws_size;
  const float* x_t       = (const float*)d_in[0];
  const float* t_in      = (const float*)d_in[1];
  const float* x0        = (const float*)d_in[2];
  const float* pc_raw    = (const float*)d_in[3];
  const float* pc_proj_w = (const float*)d_in[4];
  const float* pc_proj_b = (const float*)d_in[5];
  const float* joint_w   = (const float*)d_in[6];
  const float* joint_b   = (const float*)d_in[7];
  const float* coarse_w  = (const float*)d_in[8];
  const float* coarse_b  = (const float*)d_in[9];
  const float* diff_w    = (const float*)d_in[10];
  const float* diff_b    = (const float*)d_in[11];
  const float* id_embed  = (const float*)d_in[12];
  const float* dop_w1    = (const float*)d_in[13];
  const float* dop_b1    = (const float*)d_in[14];
  const float* dop_w2    = (const float*)d_in[15];
  const float* dop_b2    = (const float*)d_in[16];
  const float* time_w1   = (const float*)d_in[17];
  const float* time_b1   = (const float*)d_in[18];
  const float* time_w2   = (const float*)d_in[19];
  const float* time_b2   = (const float*)d_in[20];
  const float* dproj_w   = (const float*)d_in[21];
  const float* dproj_b   = (const float*)d_in[22];
  const float* sa_qkv_w  = (const float*)d_in[23];
  const float* sa_qkv_b  = (const float*)d_in[24];
  const float* sa_out_w  = (const float*)d_in[25];
  const float* sa_out_b  = (const float*)d_in[26];
  const float* ca_qkv_w  = (const float*)d_in[27];
  const float* ca_qkv_b  = (const float*)d_in[28];
  const float* ca_out_w  = (const float*)d_in[29];
  const float* ca_out_b  = (const float*)d_in[30];
  const float* ln_w      = (const float*)d_in[31];
  const float* ln_b      = (const float*)d_in[32];
  const float* ffn_w1    = (const float*)d_in[33];
  const float* ffn_b1    = (const float*)d_in[34];
  const float* ffn_w2    = (const float*)d_in[35];
  const float* ffn_b2    = (const float*)d_in[36];
  const float* vel_w     = (const float*)d_in[37];
  const float* vel_b     = (const float*)d_in[38];
  float* out = (float*)d_out;

  char* base = (char*)d_ws;
  size_t off = 0;
  auto alloc = [&](size_t bytes) -> char* {
    char* p = base + off;
    off = (off + bytes + 255) & ~((size_t)255);
    return p;
  };
  // bf16 weights
  __bf16* dproj_wb  = (__bf16*)alloc((size_t)512 * 1024 * 2);
  __bf16* sa_qkv_wb = (__bf16*)alloc((size_t)NLAY * 1536 * 512 * 2);
  __bf16* sa_out_wb = (__bf16*)alloc((size_t)NLAY * 512 * 512 * 2);
  __bf16* ca_qkv_wb = (__bf16*)alloc((size_t)NLAY * 1536 * 512 * 2);
  __bf16* ca_out_wb = (__bf16*)alloc((size_t)NLAY * 512 * 512 * 2);
  __bf16* ffn1_wb   = (__bf16*)alloc((size_t)NLAY * 2048 * 512 * 2);
  __bf16* ffn2_wb   = (__bf16*)alloc((size_t)NLAY * 512 * 2048 * 2);
  // activations / intermediates
  __bf16* mem_b   = (__bf16*)alloc((size_t)BT * NPTS * DD * 2);
  __bf16* dophT   = (__bf16*)alloc((size_t)BT * DD * NPTS * 2);
  __bf16* dw_b    = (__bf16*)alloc((size_t)BT * NJ * NPTS * 2);
  float*  maskB   = (float*) alloc((size_t)BT * NJ * NPTS * 4);
  float*  invDen  = (float*) alloc((size_t)BT * NJ * 4);
  float*  imp     = (float*) alloc((size_t)BT * NPTS * 4);
  float*  hdop    = (float*) alloc((size_t)NTOK * DD * 4);
  float*  qacc    = (float*) alloc((size_t)NTOK * DD * 4);
  float*  ht      = (float*) alloc((size_t)BT * DD * 4);
  __bf16* cc      = (__bf16*)alloc((size_t)NTOK * 1024 * 2);
  float*  xres    = (float*) alloc((size_t)NTOK * DD * 4);
  __bf16* xln     = (__bf16*)alloc((size_t)NTOK * DD * 2);
  float*  qkv     = (float*) alloc((size_t)NTOK * 1536 * 4);
  __bf16* ob      = (__bf16*)alloc((size_t)NTOK * DD * 2);
  __bf16* qb      = (__bf16*)alloc((size_t)NTOK * DD * 2);
  __bf16* kbuf    = (__bf16*)alloc((size_t)BT * NPTS * DD * 2);
  __bf16* vtbuf   = (__bf16*)alloc((size_t)BT * DD * NPTS * 2);
  __bf16* h1      = (__bf16*)alloc((size_t)NTOK * NFF * 2);

  auto cvt = [&](const float* s, __bf16* d, long long n) {
    f32_to_bf16_kernel<<<2048, 256, 0, stream>>>(s, d, n);
  };
  cvt(dproj_w,  dproj_wb,  (long long)512 * 1024);
  cvt(sa_qkv_w, sa_qkv_wb, (long long)NLAY * 1536 * 512);
  cvt(sa_out_w, sa_out_wb, (long long)NLAY * 512 * 512);
  cvt(ca_qkv_w, ca_qkv_wb, (long long)NLAY * 1536 * 512);
  cvt(ca_out_w, ca_out_wb, (long long)NLAY * 512 * 512);
  cvt(ffn_w1,   ffn1_wb,   (long long)NLAY * 2048 * 512);
  cvt(ffn_w2,   ffn2_wb,   (long long)NLAY * 512 * 2048);

  // preprocessing
  dw_mask_kernel<<<BT * NJ, 256, 0, stream>>>(x_t, pc_raw, dw_b, maskB, invDen);
  imp_kernel<<<(BT * NPTS + 255) / 256, 256, 0, stream>>>(x_t, pc_raw, imp);
  memproj_kernel<<<BT * NPTS, 128, 0, stream>>>(pc_raw, pc_proj_w, pc_proj_b, imp, mem_b);
  dop_kernel<<<BT * NPTS, 64, 0, stream>>>(pc_raw, dop_w1, dop_b1, dop_w2, dop_b2, dophT);

  // h_dop = (dw @ dop_h) / denom   (batched WMMA GEMM, EPI5)
  gemm_bf16_kernel<5><<<dim3(1, DD / 64, BT), 256, 0, stream>>>(
      dw_b, NPTS, (long long)NJ * NPTS,
      dophT, NPTS, (long long)DD * NPTS,
      nullptr, hdop, (long long)NJ * DD, nullptr, 0, nullptr,
      invDen, NJ, NJ, DD, NPTS, DD);

  jointproj_kernel<<<NTOK, 128, 0, stream>>>(x_t, x0, joint_w, joint_b, coarse_w, coarse_b,
                                             diff_w, diff_b, id_embed, cc, qacc);
  time_kernel<<<BT, 256, 0, stream>>>(t_in, time_w1, time_b1, time_w2, time_b2, ht);

  // qacc += concat(h_xt, h_parent) @ dproj^T + b   (EPI1 accumulate)
  gemm_bf16_kernel<1><<<dim3(NTOK / 128, DD / 64, 1), 256, 0, stream>>>(
      cc, 1024, 0, dproj_wb, 1024, 0, dproj_b,
      qacc, 0, nullptr, 0, nullptr, nullptr, 0, NTOK, DD, 1024, DD);

  assemble_kernel<<<(NTOK * DD) / 256, 256, 0, stream>>>(qacc, hdop, ht, xres);

  const size_t caShmem = 32 * 1024 * 4 + 32 * 1024 * 2 + 32 * 8 * 4; // 197632 B

  for (int i = 0; i < NLAY; ++i) {
    // --- self-attention block ---
    ln_kernel<<<NTOK / 8, 256, 0, stream>>>(xres, ln_w + ((long long)i * 3 + 0) * DD,
                                            ln_b + ((long long)i * 3 + 0) * DD, xln, NTOK);
    gemm_bf16_kernel<0><<<dim3(NTOK / 128, 1536 / 64, 1), 256, 0, stream>>>(
        xln, DD, 0, sa_qkv_wb + (long long)i * 1536 * 512, DD, 0, sa_qkv_b + (long long)i * 1536,
        qkv, 0, nullptr, 0, nullptr, nullptr, 0, NTOK, 1536, DD, 1536);
    sa_attn_kernel<<<BT * NHD, 32, 0, stream>>>(qkv, ob);
    gemm_bf16_kernel<1><<<dim3(NTOK / 128, DD / 64, 1), 256, 0, stream>>>(
        ob, DD, 0, sa_out_wb + (long long)i * 512 * 512, DD, 0, sa_out_b + (long long)i * DD,
        xres, 0, nullptr, 0, nullptr, nullptr, 0, NTOK, DD, DD, DD);

    // --- cross-attention block ---
    ln_kernel<<<NTOK / 8, 256, 0, stream>>>(xres, ln_w + ((long long)i * 3 + 1) * DD,
                                            ln_b + ((long long)i * 3 + 1) * DD, xln, NTOK);
    gemm_bf16_kernel<2><<<dim3(NTOK / 128, DD / 64, 1), 256, 0, stream>>>(
        xln, DD, 0, ca_qkv_wb + (long long)i * 1536 * 512, DD, 0, ca_qkv_b + (long long)i * 1536,
        nullptr, 0, qb, 0, nullptr, nullptr, 0, NTOK, DD, DD, DD);
    gemm_bf16_kernel<4><<<dim3((BT * NPTS) / 128, 1024 / 64, 1), 256, 0, stream>>>(
        mem_b, DD, 0, ca_qkv_wb + (long long)i * 1536 * 512 + (long long)512 * 512, DD, 0,
        ca_qkv_b + (long long)i * 1536 + 512,
        nullptr, 0, kbuf, 0, vtbuf, nullptr, 0, BT * NPTS, 1024, DD, 1024);
    ca_attn_kernel<<<BT * NHD, 256, caShmem, stream>>>(qb, kbuf, vtbuf, maskB, ob);
    gemm_bf16_kernel<1><<<dim3(NTOK / 128, DD / 64, 1), 256, 0, stream>>>(
        ob, DD, 0, ca_out_wb + (long long)i * 512 * 512, DD, 0, ca_out_b + (long long)i * DD,
        xres, 0, nullptr, 0, nullptr, nullptr, 0, NTOK, DD, DD, DD);

    // --- FFN block ---
    ln_kernel<<<NTOK / 8, 256, 0, stream>>>(xres, ln_w + ((long long)i * 3 + 2) * DD,
                                            ln_b + ((long long)i * 3 + 2) * DD, xln, NTOK);
    gemm_bf16_kernel<3><<<dim3(NTOK / 128, NFF / 64, 1), 256, 0, stream>>>(
        xln, DD, 0, ffn1_wb + (long long)i * 2048 * 512, DD, 0, ffn_b1 + (long long)i * NFF,
        nullptr, 0, h1, 0, nullptr, nullptr, 0, NTOK, NFF, DD, NFF);
    gemm_bf16_kernel<1><<<dim3(NTOK / 128, DD / 64, 1), 256, 0, stream>>>(
        h1, NFF, 0, ffn2_wb + (long long)i * 512 * 2048, NFF, 0, ffn_b2 + (long long)i * DD,
        xres, 0, nullptr, 0, nullptr, nullptr, 0, NTOK, DD, NFF, DD);
  }

  vel_kernel<<<(NTOK * 3 * 32 + 127) / 128, 128, 0, stream>>>(xres, vel_w, vel_b, out);
}
